// KDEResidualEstimator_85349590106796
// MI455X (gfx1250) — matrix operand order repaired and checked by reading, not compiled, not run on hardware
//
#include <hip/hip_runtime.h>

// Problem constants (match the reference exactly).
#define KM 16384   // rows of x / outputs
#define KN 16384   // rows of X / residuals
#define KD 256     // feature dim

// GEMM tiling
#define TM 128     // block rows (x)
#define TN 64      // tile cols (X) per iteration
#define LDA 264    // LDS row stride in bf16 elements (256 + 8 pad -> conflict-min)
#define SPLITN 8   // grid.y splits over N for parallelism
// fallback: single-buffered B
#define SMEM_FALLBACK ((2 * TM + 2 * TN) * LDA * 2)        // 202752 B
// async: double-buffered B for DMA/compute overlap
#define SMEM_ASYNC ((2 * TM + 4 * TN) * LDA * 2)           // 270336 B (< 320KB WGP LDS)

// workspace: [floats: inv_bw(64) colstd(->512) xnorm KM Xnorm KN num KM den KM] then bf16 arrays
#define WS_FLOATS (512 + 4 * KM)                   // 66048 floats = 264192 bytes
#define WS_BF_OFF ((size_t)WS_FLOATS * 4)          // byte offset of bf16 region (16-aligned)
#define WS_NEED (WS_BF_OFF + (size_t)(KM + KN) * KD * 2 * 2)  // + hi/lo bf16 for x and X

#if __has_builtin(__builtin_amdgcn_global_load_async_to_lds_b128) && \
    __has_builtin(__builtin_amdgcn_s_wait_asynccnt)
#define HAVE_ASYNC 1
#else
#define HAVE_ASYNC 0
#endif

typedef __attribute__((ext_vector_type(16))) __bf16 v16bf;
typedef __attribute__((ext_vector_type(8)))  __bf16 v8bf;
typedef __attribute__((ext_vector_type(8)))  float  v8f;
// match the async builtin's parameter type exactly: GCC-style vector of 4 ints
typedef int v4i_b128 __attribute__((vector_size(16)));
typedef __attribute__((address_space(1))) v4i_b128* g_b128_p;
typedef __attribute__((address_space(3))) v4i_b128* l_b128_p;

// ---------- bf16 split helpers (round-to-nearest-even) ----------
static __device__ __forceinline__ unsigned short bf16_bits(float f) {
  unsigned u = __builtin_bit_cast(unsigned, f);
  u += 0x7fffu + ((u >> 16) & 1u);
  return (unsigned short)(u >> 16);
}
static __device__ __forceinline__ float bf16_val(unsigned short h) {
  unsigned u = ((unsigned)h) << 16;
  return __builtin_bit_cast(float, u);
}
// split 4 floats into hi/lo bf16, packed, store 8B each
static __device__ __forceinline__ void split_store4(float4 f, unsigned short* hp,
                                                    unsigned short* lp) {
  unsigned short h0 = bf16_bits(f.x), h1 = bf16_bits(f.y);
  unsigned short h2 = bf16_bits(f.z), h3 = bf16_bits(f.w);
  unsigned short l0 = bf16_bits(f.x - bf16_val(h0));
  unsigned short l1 = bf16_bits(f.y - bf16_val(h1));
  unsigned short l2 = bf16_bits(f.z - bf16_val(h2));
  unsigned short l3 = bf16_bits(f.w - bf16_val(h3));
  uint2 hv, lv;
  hv.x = (unsigned)h0 | ((unsigned)h1 << 16);
  hv.y = (unsigned)h2 | ((unsigned)h3 << 16);
  lv.x = (unsigned)l0 | ((unsigned)l1 << 16);
  lv.y = (unsigned)l2 | ((unsigned)l3 << 16);
  *(uint2*)hp = hv;
  *(uint2*)lp = lv;
}

// concat two 16B LDS reads into one v16bf fragment
static __device__ __forceinline__ v16bf frag16(const unsigned short* p0,
                                               const unsigned short* p1) {
  v8bf a = *(const v8bf*)p0;
  v8bf b = *(const v8bf*)p1;
  return __builtin_shufflevector(a, b, 0, 1, 2, 3, 4, 5, 6, 7, 8, 9, 10, 11, 12, 13, 14, 15);
}

#if HAVE_ASYNC
// 16B global->LDS async copy (ASYNCcnt-tracked, no VGPR data)
static __device__ __forceinline__ void async_copy16(const unsigned short* g,
                                                    unsigned short* l) {
  __builtin_amdgcn_global_load_async_to_lds_b128((g_b128_p)g, (l_b128_p)l, 0, 0);
}
// stage one 64x256 bf16 tile pair (hi+lo) via async DMA; 16 chunks/thread
static __device__ __forceinline__ void stage_B_async(const unsigned short* __restrict__ Xh,
                                                     const unsigned short* __restrict__ Xl,
                                                     unsigned short* BhP,
                                                     unsigned short* BlP, int c0,
                                                     int tid) {
#pragma unroll
  for (int k = 0; k < 8; ++k) {
    const int c = tid + k * 256;              // 0..2047
    const int row = c >> 5;
    const int col = (c & 31) * 8;             // bf16 elements
    async_copy16(Xh + (size_t)(c0 + row) * KD + col, BhP + row * LDA + col);
    async_copy16(Xl + (size_t)(c0 + row) * KD + col, BlP + row * LDA + col);
  }
}
#endif

// ---------- preamble kernels ----------
__global__ __launch_bounds__(256) void kde_col_stats(const float* __restrict__ X,
                                                     float* __restrict__ colstd) {
  const int j = blockIdx.x;
  float s = 0.f, ss = 0.f;
  for (int i = threadIdx.x; i < KN; i += 256) {
    float v = X[(size_t)i * KD + j];
    s += v;
    ss += v * v;
  }
  __shared__ float sh0[256], sh1[256];
  sh0[threadIdx.x] = s;
  sh1[threadIdx.x] = ss;
  __syncthreads();
  for (int off = 128; off > 0; off >>= 1) {
    if (threadIdx.x < off) {
      sh0[threadIdx.x] += sh0[threadIdx.x + off];
      sh1[threadIdx.x] += sh1[threadIdx.x + off];
    }
    __syncthreads();
  }
  if (threadIdx.x == 0) {
    float sum = sh0[0], sumsq = sh1[0];
    float var = (sumsq - sum * sum / (float)KN) / (float)(KN - 1);
    colstd[j] = sqrtf(fmaxf(var, 0.f));
  }
}

__global__ __launch_bounds__(256) void kde_finalize_h(const float* __restrict__ colstd,
                                                      float* __restrict__ inv_bw) {
  __shared__ float sh[256];
  sh[threadIdx.x] = colstd[threadIdx.x];
  __syncthreads();
  for (int off = 128; off > 0; off >>= 1) {
    if (threadIdx.x < off) sh[threadIdx.x] += sh[threadIdx.x + off];
    __syncthreads();
  }
  if (threadIdx.x == 0) {
    float mean = sh[0] / (float)KD;
    float h = mean * expf(-logf((float)KN) / (float)(KD + 4));
    inv_bw[0] = 1.0f / (2.0f * h * h);
  }
}

__global__ __launch_bounds__(256) void kde_row_norms(const float* __restrict__ x,
                                                     const float* __restrict__ X,
                                                     float* __restrict__ xn,
                                                     float* __restrict__ Xn) {
  const int wave = threadIdx.x >> 5;
  const int lane = threadIdx.x & 31;
  const int row = blockIdx.x * 8 + wave;
  const float* src;
  float* dst;
  if (row < KM) {
    src = x + (size_t)row * KD;
    dst = xn + row;
  } else {
    src = X + (size_t)(row - KM) * KD;
    dst = Xn + (row - KM);
  }
  float s = 0.f;
#pragma unroll
  for (int k = 0; k < KD / 32; ++k) {
    float v = src[lane + k * 32];
    s += v * v;
  }
#pragma unroll
  for (int m = 16; m >= 1; m >>= 1) s += __shfl_xor(s, m, 32);
  if (lane == 0) *dst = s;
}

__global__ __launch_bounds__(256) void kde_zero(float* __restrict__ p, int n) {
  int i = blockIdx.x * 256 + threadIdx.x;
  if (i < n) p[i] = 0.f;
}

__global__ __launch_bounds__(256) void kde_finalize_out(const float* __restrict__ num,
                                                        const float* __restrict__ den,
                                                        float* __restrict__ out) {
  int i = blockIdx.x * 256 + threadIdx.x;
  if (i < KM) out[i] = num[i] / (den[i] + 1e-8f);
}

// one-time fp32 -> bf16 hi/lo pre-conversion of x and X into workspace
__global__ __launch_bounds__(256) void kde_convert(const float* __restrict__ x,
                                                   const float* __restrict__ X,
                                                   unsigned short* __restrict__ xh,
                                                   unsigned short* __restrict__ xl,
                                                   unsigned short* __restrict__ Xh,
                                                   unsigned short* __restrict__ Xl) {
  const size_t i = (size_t)blockIdx.x * 256 + threadIdx.x;  // index of a 4-float chunk
  const size_t nx = (size_t)KM * KD / 4;
  float4 f;
  unsigned short *hp, *lp;
  if (i < nx) {
    f = ((const float4*)x)[i];
    hp = xh + i * 4;
    lp = xl + i * 4;
  } else {
    const size_t j = i - nx;
    f = ((const float4*)X)[j];
    hp = Xh + j * 4;
    lp = Xl + j * 4;
  }
  split_store4(f, hp, lp);
}

// ---------- shared GEMM+epilogue body for one 128x64 tile ----------
static __device__ __forceinline__ void kde_compute_tile(
    const unsigned short* __restrict__ a_h, const unsigned short* __restrict__ a_l,
    const unsigned short* __restrict__ Bh, const unsigned short* __restrict__ Bl,
    int l15, int akoff, int bkoff, int c0, const float* __restrict__ res,
    const float* __restrict__ Xnorm, float inv_bw, const float* xn, float* numv,
    float* denv) {
  v8f acc[4];
#pragma unroll
  for (int t = 0; t < 4; ++t)
#pragma unroll
    for (int e = 0; e < 8; ++e) acc[t][e] = 0.f;

#pragma unroll
  for (int kc = 0; kc < KD; kc += 32) {
    v16bf ahi = frag16(a_h + kc + akoff, a_h + kc + akoff + 16);
    v16bf alo = frag16(a_l + kc + akoff, a_l + kc + akoff + 16);
#pragma unroll
    for (int t = 0; t < 4; ++t) {
      const unsigned short* pbh = Bh + (t * 16 + l15) * LDA + kc + bkoff;
      const unsigned short* pbl = Bl + (t * 16 + l15) * LDA + kc + bkoff;
      v16bf bhi = frag16(pbh, pbh + 16);
      v16bf blo = frag16(pbl, pbl + 16);
      acc[t] = __builtin_amdgcn_wmma_f32_16x16x32_bf16(false, ahi, false, bhi, (short)0,
                                                       acc[t], false, false);
      acc[t] = __builtin_amdgcn_wmma_f32_16x16x32_bf16(false, ahi, false, blo, (short)0,
                                                       acc[t], false, false);
      acc[t] = __builtin_amdgcn_wmma_f32_16x16x32_bf16(false, alo, false, bhi, (short)0,
                                                       acc[t], false, false);
    }
  }

  // fused epilogue: w = exp(-(||x||^2 + ||X||^2 - 2 x.X) * inv_bw)
#pragma unroll
  for (int t = 0; t < 4; ++t) {
    const int col = c0 + t * 16 + l15;
    const float rv = res[col];
    const float Xnv = Xnorm[col];
#pragma unroll
    for (int v = 0; v < 8; ++v) {
      float sq = xn[v] + Xnv - 2.0f * acc[t][v];
      sq = fmaxf(sq, 0.f);
      float wgt = __expf(-sq * inv_bw);
      numv[v] += wgt * rv;
      denv[v] += wgt;
    }
  }
}

static __device__ __forceinline__ void kde_reduce_store(int r0, int wave, int lane,
                                                        const float* numv,
                                                        const float* denv,
                                                        float* __restrict__ g_num,
                                                        float* __restrict__ g_den) {
  const int hl = lane >> 4;
  const int l15 = lane & 15;
#pragma unroll
  for (int v = 0; v < 8; ++v) {
    float nv = numv[v], dv = denv[v];
#pragma unroll
    for (int m = 8; m >= 1; m >>= 1) {
      nv += __shfl_xor(nv, m, 16);
      dv += __shfl_xor(dv, m, 16);
    }
    if (l15 == 0) {
      const int row = r0 + wave * 16 + hl * 8 + v;
      atomicAdd(&g_num[row], nv);
      atomicAdd(&g_den[row], dv);
    }
  }
}

#if HAVE_ASYNC
// ---------- main kernel, async path: pre-converted bf16 + double-buffered DMA ----------
__global__ __launch_bounds__(256) void kde_main_async(
    const unsigned short* __restrict__ xh, const unsigned short* __restrict__ xl,
    const unsigned short* __restrict__ Xh, const unsigned short* __restrict__ Xl,
    const float* __restrict__ res, const float* __restrict__ xnorm,
    const float* __restrict__ Xnorm, const float* __restrict__ inv_bw_p,
    float* __restrict__ g_num, float* __restrict__ g_den) {
  extern __shared__ unsigned short smem[];
  unsigned short* Ah = smem;                       // TM x LDA
  unsigned short* Al = Ah + TM * LDA;              // TM x LDA
  unsigned short* Bbase = Al + TM * LDA;           // 2 buffers x (Bh,Bl) of TN x LDA
  const int BSPAN = 2 * TN * LDA;                  // one (hi+lo) B buffer

  const int tid = threadIdx.x;
  const int wave = tid >> 5;
  const int lane = tid & 31;
  const int hl = lane >> 4;
  const int l15 = lane & 15;

  const int r0 = blockIdx.x * TM;
  const int nchunk = KN / (int)gridDim.y;
  const int c_begin = blockIdx.y * nchunk;
  const int c_end = c_begin + nchunk;

  // prologue: async-stage A tile (hi+lo) and the first B tile into buffer 0
  {
    const unsigned short* gh = xh + (size_t)r0 * KD;
    const unsigned short* gl = xl + (size_t)r0 * KD;
#pragma unroll
    for (int k = 0; k < 16; ++k) {
      const int c = tid + k * 256;              // 0..4095
      const int row = c >> 5;
      const int col = (c & 31) * 8;             // bf16 elements
      async_copy16(gh + row * KD + col, Ah + row * LDA + col);
      async_copy16(gl + row * KD + col, Al + row * LDA + col);
    }
  }
  stage_B_async(Xh, Xl, Bbase, Bbase + TN * LDA, c_begin, tid);

  const float inv_bw = inv_bw_p[0];
  float xn[8];
  {
    const int rbase = r0 + wave * 16 + hl * 8;
#pragma unroll
    for (int v = 0; v < 8; ++v) xn[v] = xnorm[rbase + v];
  }
  float numv[8], denv[8];
#pragma unroll
  for (int v = 0; v < 8; ++v) { numv[v] = 0.f; denv[v] = 0.f; }

  const unsigned short* a_h = Ah + (wave * 16 + l15) * LDA;
  const unsigned short* a_l = Al + (wave * 16 + l15) * LDA;
  const int akoff = hl * 8;
  const int bkoff = hl * 8;

  __builtin_amdgcn_s_wait_asynccnt(0);  // A + first B resident (this wave's share)
  __syncthreads();                      // ... and everyone else's

  int p = 0;
  for (int c0 = c_begin; c0 < c_end; c0 += TN) {
    // prefetch next B tile into the other buffer while we compute on this one;
    // buffer 1-p was last read in the previous iteration, already fenced by the
    // wait+barrier at the end of that iteration.
    if (c0 + TN < c_end)
      stage_B_async(Xh, Xl, Bbase + (1 - p) * BSPAN, Bbase + (1 - p) * BSPAN + TN * LDA,
                    c0 + TN, tid);

    kde_compute_tile(a_h, a_l, Bbase + p * BSPAN, Bbase + p * BSPAN + TN * LDA, l15,
                     akoff, bkoff, c0, res, Xnorm, inv_bw, xn, numv, denv);

    __builtin_amdgcn_s_wait_asynccnt(0);  // next tile's DMA done (overlapped w/ compute)
    __syncthreads();                      // all waves: done computing + done copying
    p ^= 1;
  }
  kde_reduce_store(r0, wave, lane, numv, denv, g_num, g_den);
}
#endif  // HAVE_ASYNC

// ---------- main kernel, fallback path (in-kernel fp32 -> bf16 split) ----------
__global__ __launch_bounds__(256) void kde_main_fallback(
    const float* __restrict__ x, const float* __restrict__ X,
    const float* __restrict__ res, const float* __restrict__ xnorm,
    const float* __restrict__ Xnorm, const float* __restrict__ inv_bw_p,
    float* __restrict__ g_num, float* __restrict__ g_den) {
  extern __shared__ unsigned short smem[];
  unsigned short* Ah = smem;
  unsigned short* Al = Ah + TM * LDA;
  unsigned short* Bh = Al + TM * LDA;
  unsigned short* Bl = Bh + TN * LDA;

  const int tid = threadIdx.x;
  const int wave = tid >> 5;
  const int lane = tid & 31;
  const int hl = lane >> 4;
  const int l15 = lane & 15;

  const int r0 = blockIdx.x * TM;
  const int nchunk = KN / (int)gridDim.y;
  const int c_begin = blockIdx.y * nchunk;

  {
    const int row = tid >> 1;
    const int cb = (tid & 1) * 128;
    const float4* g = (const float4*)(x + (size_t)(r0 + row) * KD + cb);
    unsigned short* ah = Ah + row * LDA + cb;
    unsigned short* al = Al + row * LDA + cb;
#pragma unroll
    for (int i = 0; i < 32; ++i) split_store4(g[i], ah + i * 4, al + i * 4);
  }

  const float inv_bw = inv_bw_p[0];
  float xn[8];
  {
    const int rbase = r0 + wave * 16 + hl * 8;
#pragma unroll
    for (int v = 0; v < 8; ++v) xn[v] = xnorm[rbase + v];
  }
  float numv[8], denv[8];
#pragma unroll
  for (int v = 0; v < 8; ++v) { numv[v] = 0.f; denv[v] = 0.f; }

  const unsigned short* a_h = Ah + (wave * 16 + l15) * LDA;
  const unsigned short* a_l = Al + (wave * 16 + l15) * LDA;
  const int akoff = hl * 8;
  const int bkoff = hl * 8;

  for (int c0 = c_begin; c0 < c_begin + nchunk; c0 += TN) {
    __syncthreads();
    {
      const int row = tid >> 2;
      const int cb = (tid & 3) * 64;
      const float4* g = (const float4*)(X + (size_t)(c0 + row) * KD + cb);
      unsigned short* bh = Bh + row * LDA + cb;
      unsigned short* bl = Bl + row * LDA + cb;
#pragma unroll
      for (int i = 0; i < 16; ++i) split_store4(g[i], bh + i * 4, bl + i * 4);
    }
    __syncthreads();

    kde_compute_tile(a_h, a_l, Bh, Bl, l15, akoff, bkoff, c0, res, Xnorm, inv_bw, xn,
                     numv, denv);
  }
  kde_reduce_store(r0, wave, lane, numv, denv, g_num, g_den);
}

// ---------- host launcher ----------
extern "C" void kernel_launch(void* const* d_in, const int* in_sizes, int n_in,
                              void* d_out, int out_size, void* d_ws, size_t ws_size,
                              hipStream_t stream) {
  const float* x = (const float*)d_in[0];     // (M, D) f32
  const float* X = (const float*)d_in[1];     // (N, D) f32
  const float* res = (const float*)d_in[2];   // (N,)  f32
  float* out = (float*)d_out;                 // (M,)  f32

  float* wsf = (float*)d_ws;
  float* inv_bw = wsf;
  float* colstd = wsf + 64;
  float* xnorm = wsf + 512;
  float* Xnorm = xnorm + KM;
  float* g_num = Xnorm + KN;
  float* g_den = g_num + KM;

  unsigned short* bfbase = (unsigned short*)((char*)d_ws + WS_BF_OFF);
  unsigned short* xh = bfbase;
  unsigned short* xl = xh + (size_t)KM * KD;
  unsigned short* Xh = xl + (size_t)KM * KD;
  unsigned short* Xl = Xh + (size_t)KN * KD;

  kde_col_stats<<<KD, 256, 0, stream>>>(X, colstd);
  kde_finalize_h<<<1, 256, 0, stream>>>(colstd, inv_bw);
  kde_row_norms<<<(KM + KN) / 8, 256, 0, stream>>>(x, X, xnorm, Xnorm);
  kde_zero<<<(2 * KM + 255) / 256, 256, 0, stream>>>(g_num, 2 * KM);

  dim3 grid(KM / TM, SPLITN);
#if HAVE_ASYNC
  if (ws_size >= WS_NEED) {
    kde_convert<<<(int)((size_t)(KM + KN) * KD / 4 / 256), 256, 0, stream>>>(x, X, xh, xl,
                                                                             Xh, Xl);
    kde_main_async<<<grid, 256, SMEM_ASYNC, stream>>>(xh, xl, Xh, Xl, res, xnorm, Xnorm,
                                                      inv_bw, g_num, g_den);
  } else
#endif
  {
    (void)xh; (void)xl; (void)Xh; (void)Xl;
    kde_main_fallback<<<grid, 256, SMEM_FALLBACK, stream>>>(x, X, res, xnorm, Xnorm,
                                                            inv_bw, g_num, g_den);
  }

  kde_finalize_out<<<(KM + 255) / 256, 256, 0, stream>>>(g_num, g_den, out);
}
